// RSSM_86955907875499
// MI455X (gfx1250) — matrix operand order repaired and checked by reading, not compiled, not run on hardware
//
#include <hip/hip_runtime.h>

// ---------------------------------------------------------------------------
// RSSM temporal scan for MI455X (gfx1250, wave32, WMMA bf16 16x16x32).
//
//   * B=256 batch rows -> 16 independent workgroups, each owns an M=16 batch
//     tile and runs the whole T=512 recurrence (no inter-WG dependence).
//   * Block = 416 threads = 13 wave32s: exactly one 16-wide output n-tile per
//     wave in every GEMM phase -> shortest serial WMMA chain per step.
//   * Weights pre-packed once into exact 32x16 WMMA B-fragment layout (bf16)
//     in workspace: one fragment = 2 contiguous global_load_b128 from L2.
//     A per-iteration opaque zero offset (volatile s_mov asm) keeps the
//     loop-invariant fragment loads inside the t-loop (no scratch spills)
//     while preserving global address-space provenance (no flat_load).
//   * Activations (x=[z|act], [h|obs], hidden) in LDS bf16; A fragment =
//     2 ds_load_b128 (ISA 7.12.2 layout).
//   * GRU gates / ELU / softplus evaluated in-register on C-fragment layout.
//   * Next-step obs/act prefetched (global_prefetch) to hide latency across
//     the serial scan.
// ---------------------------------------------------------------------------

typedef __attribute__((ext_vector_type(16))) __bf16 v16bf;
typedef __attribute__((ext_vector_type(8)))  float  v8f;

// ---- problem dims ----------------------------------------------------------
constexpr int H = 200, Z = 30, A = 128, E = 256, HID = 200;
constexpr int B = 256, T = 512;

// ---- padded dims -----------------------------------------------------------
constexpr int XK   = 160;  // [z(30)+pad2 | act(128)]  -> 5 k-tiles of 32
constexpr int AK   = 480;  // [h(200)+pad24 | obs(256)]-> 15 k-tiles
constexpr int HIDK = 224;  // hidden as K              -> 7 k-tiles
constexpr int KT_X = 5, KT_H = 7, KT_A = 15, KT_HID = 7;
constexpr int NT_G = 13;   // 208 output cols (gates / hidden)
constexpr int NT_O = 4;    // 64 output cols (2Z heads)
constexpr int NTHREADS = 416;  // 13 waves

// ---- packed-fragment workspace layout (bf16 elements) ----------------------
constexpr size_t FR      = 512;                        // elems per 32x16 tile
constexpr size_t SZ_WIH  = (size_t)KT_X   * NT_G * FR;
constexpr size_t SZ_WHH  = (size_t)KT_H   * NT_G * FR;
constexpr size_t SZ_PW1  = (size_t)KT_HID * NT_G * FR;
constexpr size_t SZ_PW2  = (size_t)KT_HID * NT_O * FR;
constexpr size_t SZ_QW1  = (size_t)KT_A   * NT_G * FR;
constexpr size_t O_WIHR = 0;
constexpr size_t O_WIHZ = O_WIHR + SZ_WIH;
constexpr size_t O_WIHN = O_WIHZ + SZ_WIH;
constexpr size_t O_WHHR = O_WIHN + SZ_WIH;
constexpr size_t O_WHHZ = O_WHHR + SZ_WHH;
constexpr size_t O_WHHN = O_WHHZ + SZ_WHH;
constexpr size_t O_PW1  = O_WHHN + SZ_WHH;
constexpr size_t O_PW2  = O_PW1  + SZ_PW1;
constexpr size_t O_QW1  = O_PW2  + SZ_PW2;
constexpr size_t O_QW2  = O_QW1  + SZ_QW1;

// ---- device helpers --------------------------------------------------------
__device__ __forceinline__ v8f zero_v8f() {
  v8f z = {0.f, 0.f, 0.f, 0.f, 0.f, 0.f, 0.f, 0.f};
  return z;
}

__device__ __forceinline__ v8f wmma_bf16(v16bf a, v16bf b, v8f c) {
  return __builtin_amdgcn_wmma_f32_16x16x32_bf16(
      /*neg_a=*/false, a, /*neg_b=*/false, b,
      /*c_mod=*/(short)0, c, /*reuse_a=*/false, /*reuse_b=*/false);
}

// A-matrix 16x32 bf16 fragment from LDS (ISA 7.12.2 layout):
//   lane L (row = L&15): VGPR0..3 hold K = half*8 + {0..7},
//                        VGPR4..7 hold K = 16 + half*8 + {0..7}, half = L>>4.
__device__ __forceinline__ v16bf load_a(const __bf16* sm, int rowElems,
                                        int kbase, int lane) {
  const int half = lane >> 4, row = lane & 15;
  const __bf16* p = sm + row * rowElems + kbase + half * 8;
  union { uint4 u[2]; v16bf v; } r;
  r.u[0] = *(const uint4*)(p);
  r.u[1] = *(const uint4*)(p + 16);
  return r.v;
}

// B fragment: lane-major packed in global ws -> 32 contiguous bytes per lane.
__device__ __forceinline__ v16bf load_b(const __bf16* frag, int lane) {
  return *(const v16bf*)(frag + (size_t)lane * 16);
}

__device__ __forceinline__ float sigf(float x) { return 1.f / (1.f + expf(-x)); }
__device__ __forceinline__ float eluf(float x) { return x > 0.f ? x : (expf(x) - 1.f); }
__device__ __forceinline__ float softplusf(float x) {
  return x > 15.f ? x : log1pf(expf(x));
}

// ---------------------------------------------------------------------------
// Weight pre-pack: fp32 (N_out x K_in, PyTorch layout) -> bf16 B fragments.
// B-tile layout (32x16, K x N): lane L -> column n = L&15,
//   elem v (0..15) -> k_local = (L>>4)*8 + v + (v>=8 ? 8 : 0).
// mapmode: 0 = identity K;  1 = GRU x-input (z padded 30->32);
//          2 = posterior    (h padded 200->224, then obs 256).
// ---------------------------------------------------------------------------
__global__ void rssm_pack_kernel(const float* __restrict__ src,
                                 __bf16* __restrict__ dst,
                                 int KT, int NT, int realN, int realK,
                                 int rowOff, int srcK, int mode) {
  int tid = blockIdx.x * blockDim.x + threadIdx.x;
  int total = KT * NT * 256;
  if (tid >= total) return;
  int v    = tid & 7;
  int lane = (tid >> 3) & 31;
  int ntk  = tid >> 8;
  int nt   = ntk % NT, kt = ntk / NT;
  int n    = nt * 16 + (lane & 15);
  int klb  = (lane >> 4) * 8 + v * 2 + ((v >> 2) << 3);
  for (int e = 0; e < 2; ++e) {
    int k = kt * 32 + klb + e;
    int rk = 0; bool valid;
    if (mode == 1) {            // x = [z(30) pad2 act(128)]
      if (k < 30)      { rk = k;      valid = true; }
      else if (k < 32) {              valid = false; }
      else             { rk = k - 2;  valid = rk < 158; }
    } else if (mode == 2) {     // [h(200) pad24 obs(256)]
      if (k < 224)     { rk = k;               valid = k < 200; }
      else             { rk = 200 + (k - 224); valid = true; }
    } else {
      rk = k; valid = k < realK;
    }
    float val = (n < realN && valid) ? src[(size_t)(rowOff + n) * srcK + rk]
                                     : 0.f;
    dst[(size_t)((kt * NT + nt) * 32 + lane) * 16 + v * 2 + e] = (__bf16)val;
  }
}

// ---------------------------------------------------------------------------
// Persistent scan kernel: grid = 16 WGs (one 16-row batch tile each),
// block = 416 threads = 13 wave32s; wave w owns output n-tile w.
// ---------------------------------------------------------------------------
__global__ __launch_bounds__(NTHREADS) void rssm_scan_kernel(
    const float* __restrict__ obs,  const float* __restrict__ act,
    const float* __restrict__ noise,
    const float* __restrict__ bih,  const float* __restrict__ bhh,
    const float* __restrict__ pb1,  const float* __restrict__ pb2,
    const float* __restrict__ qb1,  const float* __restrict__ qb2,
    const __bf16* __restrict__ wsb, float* __restrict__ out) {
  __shared__ __bf16 actb[16 * AK];    // rows x [h | pad | obs]
  __shared__ __bf16 xb[16 * XK];      // rows x [z | pad | action]
  __shared__ __bf16 a1p[16 * HIDK];   // prior hidden
  __shared__ __bf16 a1q[16 * HIDK];   // posterior hidden
  __shared__ float  qbuf[16 * 64];    // qm / qs staging for rsample

  const int tid  = threadIdx.x;
  const int lane = tid & 31, wave = tid >> 5;   // wave in [0,13)
  const int half = lane >> 4, lcol = lane & 15;
  const int b0   = blockIdx.x * 16;

  float* out_h  = out;
  float* out_z  = out_h  + (size_t)B * T * H;
  float* out_pm = out_z  + (size_t)B * T * Z;
  float* out_ps = out_pm + (size_t)B * T * Z;
  float* out_qm = out_ps + (size_t)B * T * Z;
  float* out_qs = out_qm + (size_t)B * T * Z;

  // h0 = 0, z0 = 0, and all K padding columns = 0 (stay 0 forever).
  for (int i = tid; i < 16 * AK; i += NTHREADS) actb[i] = (__bf16)0.f;
  for (int i = tid; i < 16 * XK; i += NTHREADS) xb[i]   = (__bf16)0.f;
  for (int i = tid; i < 16 * HIDK; i += NTHREADS) { a1p[i] = (__bf16)0.f; a1q[i] = (__bf16)0.f; }
  __syncthreads();

  for (int t = 0; t < T; ++t) {
    // Opaque zero: re-materialized by volatile asm every iteration, so all
    // weight-fragment addresses below are formally loop-variant (no LICM
    // hoist -> no spills), while the pointer chain still derives from the
    // kernel argument -> address space stays global (global_load_b128).
    int zoff;
    asm volatile("s_mov_b32 %0, 0" : "=s"(zoff));
    const __bf16* wb = wsb + zoff;
    const __bf16* WihR = wb + O_WIHR;  const __bf16* WihZ = wb + O_WIHZ;
    const __bf16* WihN = wb + O_WIHN;  const __bf16* WhhR = wb + O_WHHR;
    const __bf16* WhhZ = wb + O_WHHZ;  const __bf16* WhhN = wb + O_WHHN;
    const __bf16* PW1  = wb + O_PW1;   const __bf16* PW2  = wb + O_PW2;
    const __bf16* QW1  = wb + O_QW1;   const __bf16* QW2  = wb + O_QW2;

    // ---- stage obs_t / act_t into LDS; prefetch next step -----------------
    for (int i = tid; i < 16 * E; i += NTHREADS) {
      int r = i >> 8, c = i & 255;
      const float* p = &obs[((size_t)(b0 + r) * T + t) * E + c];
      actb[r * AK + 224 + c] = (__bf16)(*p);
      if ((c & 31) == 0 && t + 1 < T) __builtin_prefetch(p + E, 0, 3);
    }
    for (int i = tid; i < 16 * A; i += NTHREADS) {
      int r = i >> 7, c = i & 127;
      const float* p = &act[((size_t)(b0 + r) * T + t) * A + c];
      xb[r * XK + 32 + c] = (__bf16)(*p);
      if ((c & 31) == 0 && t + 1 < T) __builtin_prefetch(p + A, 0, 3);
    }
    __syncthreads();

    // ---- GRU cell: one n-tile per wave ------------------------------------
    {
      const int nt = wave;
      v8f ar = zero_v8f(), az = zero_v8f(), axn = zero_v8f(), ahn = zero_v8f();
      for (int kt = 0; kt < KT_X; ++kt) {
        v16bf a = load_a(xb, XK, kt * 32, lane);
        size_t to = ((size_t)kt * NT_G + nt) * FR;
        ar  = wmma_bf16(a, load_b(WihR + to, lane), ar);
        az  = wmma_bf16(a, load_b(WihZ + to, lane), az);
        axn = wmma_bf16(a, load_b(WihN + to, lane), axn);
      }
      for (int kt = 0; kt < KT_H; ++kt) {
        v16bf a = load_a(actb, AK, kt * 32, lane);
        size_t to = ((size_t)kt * NT_G + nt) * FR;
        ar  = wmma_bf16(a, load_b(WhhR + to, lane), ar);
        az  = wmma_bf16(a, load_b(WhhZ + to, lane), az);
        ahn = wmma_bf16(a, load_b(WhhN + to, lane), ahn);
      }
      const int col = nt * 16 + lcol;
      const bool valid = col < H;
      float br = 0.f, bz = 0.f, bxn = 0.f, bhn = 0.f;
      if (valid) {
        br  = bih[col] + bhh[col];
        bz  = bih[200 + col] + bhh[200 + col];
        bxn = bih[400 + col];
        bhn = bhh[400 + col];
      }
      float htv[8];
#pragma unroll
      for (int i = 0; i < 8; ++i) {                  // C layout: row=i+8*half
        float hp = (float)actb[(i + 8 * half) * AK + col];
        float rg = sigf(ar[i] + br);
        float zg = sigf(az[i] + bz);
        float ng = tanhf(axn[i] + bxn + rg * (ahn[i] + bhn));
        float ht = (1.f - zg) * ng + zg * hp;
        htv[i] = ht;
        if (valid)
          out_h[((size_t)(b0 + i + 8 * half) * T + t) * H + col] = ht;
      }
      __syncthreads();                               // all h_prev reads done
      if (valid) {
#pragma unroll
        for (int i = 0; i < 8; ++i)
          actb[(i + 8 * half) * AK + col] = (__bf16)htv[i];
      }
    }
    __syncthreads();                                 // h_t visible to all

    // ---- layer 1 of both heads (prior: K=h, posterior: K=h|obs) -----------
    {
      const int nt = wave;
      v8f ap = zero_v8f();
      for (int kt = 0; kt < KT_HID; ++kt)
        ap = wmma_bf16(load_a(actb, AK, kt * 32, lane),
                       load_b(PW1 + ((size_t)kt * NT_G + nt) * FR, lane), ap);
      v8f aq = zero_v8f();
      for (int kt = 0; kt < KT_A; ++kt)
        aq = wmma_bf16(load_a(actb, AK, kt * 32, lane),
                       load_b(QW1 + ((size_t)kt * NT_G + nt) * FR, lane), aq);
      const int col = nt * 16 + lcol;
      if (col < HID) {
        float b1p = pb1[col], b1q = qb1[col];
#pragma unroll
        for (int i = 0; i < 8; ++i) {
          a1p[(i + 8 * half) * HIDK + col] = (__bf16)eluf(ap[i] + b1p);
          a1q[(i + 8 * half) * HIDK + col] = (__bf16)eluf(aq[i] + b1q);
        }
      }
    }
    __syncthreads();

    // ---- layer 2: waves 0..3 prior tiles, 4..7 posterior tiles ------------
    if (wave < 8) {
      const bool isPost = wave >= 4;
      const int  nt = wave & 3;
      const __bf16* sm = isPost ? a1q : a1p;
      const __bf16* wf = isPost ? QW2 : PW2;
      const float*  b2 = isPost ? qb2 : pb2;
      v8f acc = zero_v8f();
      for (int kt = 0; kt < KT_HID; ++kt)
        acc = wmma_bf16(load_a(sm, HIDK, kt * 32, lane),
                        load_b(wf + ((size_t)kt * NT_O + nt) * FR, lane), acc);
      int c = nt * 16 + lcol;
      if (c < 2 * Z) {
        float bb = b2[c];
#pragma unroll
        for (int i = 0; i < 8; ++i) {
          int row = i + 8 * half;
          size_t base = ((size_t)(b0 + row) * T + t) * Z;
          float v = acc[i] + bb;
          if (c < Z) {
            if (isPost) { out_qm[base + c] = v; qbuf[row * 64 + c] = v; }
            else          out_pm[base + c] = v;
          } else {
            float s = softplusf(v) + 0.1f;
            int c2 = c - Z;
            if (isPost) { out_qs[base + c2] = s; qbuf[row * 64 + 32 + c2] = s; }
            else          out_ps[base + c2] = s;
          }
        }
      }
    }
    __syncthreads();

    // ---- rsample: z_t = qm + qs * eps; feed next step's x -----------------
    for (int i = tid; i < 16 * Z; i += NTHREADS) {
      int r = i / Z, j = i - r * Z;
      float eps = noise[((size_t)(b0 + r) * T + t) * Z + j];
      float zt  = qbuf[r * 64 + j] + qbuf[r * 64 + 32 + j] * eps;
      out_z[((size_t)(b0 + r) * T + t) * Z + j] = zt;
      xb[r * XK + j] = (__bf16)zt;
    }
    __syncthreads();
  }
}

// ---------------------------------------------------------------------------
extern "C" void kernel_launch(void* const* d_in, const int* in_sizes, int n_in,
                              void* d_out, int out_size, void* d_ws,
                              size_t ws_size, hipStream_t stream) {
  (void)in_sizes; (void)n_in; (void)out_size; (void)ws_size;

  const float* obs   = (const float*)d_in[0];
  const float* actIn = (const float*)d_in[1];
  const float* noise = (const float*)d_in[2];
  const float* Wih   = (const float*)d_in[3];
  const float* Whh   = (const float*)d_in[4];
  const float* bih   = (const float*)d_in[5];
  const float* bhh   = (const float*)d_in[6];
  const float* pW1   = (const float*)d_in[7];
  const float* pb1   = (const float*)d_in[8];
  const float* pW2   = (const float*)d_in[9];
  const float* pb2   = (const float*)d_in[10];
  const float* qW1   = (const float*)d_in[11];
  const float* qb1   = (const float*)d_in[12];
  const float* qW2   = (const float*)d_in[13];
  const float* qb2   = (const float*)d_in[14];

  __bf16* wsb = (__bf16*)d_ws;
  float*  out = (float*)d_out;

  auto pack = [&](const float* src, size_t off, int KT, int NT, int realN,
                  int realK, int rowOff, int srcK, int mode) {
    int total = KT * NT * 256;
    rssm_pack_kernel<<<(total + 255) / 256, 256, 0, stream>>>(
        src, wsb + off, KT, NT, realN, realK, rowOff, srcK, mode);
  };

  // GRU input-weights (gate slices of Wih, K mapped with z->32 padding)
  pack(Wih, O_WIHR, KT_X, NT_G, H, 0, 0,   Z + A, 1);
  pack(Wih, O_WIHZ, KT_X, NT_G, H, 0, 200, Z + A, 1);
  pack(Wih, O_WIHN, KT_X, NT_G, H, 0, 400, Z + A, 1);
  // GRU hidden-weights (gate slices of Whh)
  pack(Whh, O_WHHR, KT_H, NT_G, H, H, 0,   H, 0);
  pack(Whh, O_WHHZ, KT_H, NT_G, H, H, 200, H, 0);
  pack(Whh, O_WHHN, KT_H, NT_G, H, H, 400, H, 0);
  // prior MLP
  pack(pW1, O_PW1, KT_HID, NT_G, HID,   H,   0, H,   0);
  pack(pW2, O_PW2, KT_HID, NT_O, 2 * Z, HID, 0, HID, 0);
  // posterior MLP (layer-1 K = [h|obs] with h padded to 224)
  pack(qW1, O_QW1, KT_A,   NT_G, HID,   0,   0, H + E, 2);
  pack(qW2, O_QW2, KT_HID, NT_O, 2 * Z, HID, 0, HID,   0);

  rssm_scan_kernel<<<dim3(B / 16), dim3(NTHREADS), 0, stream>>>(
      obs, actIn, noise, bih, bhh, pb1, pb2, qb1, qb2, wsb, out);
}
